// HierVQ_57466662420710
// MI455X (gfx1250) — matrix operand order repaired and checked by reading, not compiled, not run on hardware
//
#include <hip/hip_runtime.h>

typedef __attribute__((ext_vector_type(16))) _Float16 v16h;
typedef __attribute__((ext_vector_type(8)))  float    v8f;

#define CDIM   256
#define KCODES 1024
#define TLEN   2048
#define NTILES 2048          // 32768 rows / 16
#define NELEM  8388608       // B*C*T

// ---- CDNA5 async memory->LDS (ASYNCcnt) helpers -------------------
__device__ __forceinline__ unsigned lds_off(const void* p) {
  // flat LDS addresses carry the DS-usable offset in addr[31:0]
  return (unsigned)(size_t)p;
}
__device__ __forceinline__ void async_ld_b128(unsigned ldsaddr, const void* gaddr) {
  asm volatile("global_load_async_to_lds_b128 %0, %1, off"
               :: "v"(ldsaddr), "v"((unsigned long long)(size_t)gaddr)
               : "memory");
}
__device__ __forceinline__ void wait_async0() {
  asm volatile("s_wait_asynccnt 0x0" ::: "memory");
}

// ------------------------------------------------------------------
// Prep: repack emb (f32 [K,C]) into f16 B-matrix tiles for
// V_WMMA_F32_16X16X32_F16.  Per (n,kc) tile of 512 halves:
//   element[lane*16 + h] = emb[n*16 + (lane&15)][kc*32 + ((lane>>4)<<4) + h]
// contiguous k order, same as the A fragments (lanes 0-15: K=0..15, lanes 16-31: K=16..31).
__global__ __launch_bounds__(256) void hvq_prep_embT(
    const float* __restrict__ e0, const float* __restrict__ e1,
    const float* __restrict__ e2, _Float16* __restrict__ embT)
{
  int g = blockIdx.x * 256 + threadIdx.x;   // 0 .. 3*262144-1
  int l = g >> 18;
  int r = g & 262143;
  int tile = r >> 9;                         // n*8 + kc
  int eidx = r & 511;                        // lane*16 + h
  int lane = eidx >> 4, h = eidx & 15;
  int n = tile >> 3, kc = tile & 7;
  int code = n * 16 + (lane & 15);
  int c = kc * 32 + ((lane >> 4) << 4) + h;
  const float* e = (l == 0) ? e0 : ((l == 1) ? e1 : e2);
  embT[g] = (_Float16)e[code * CDIM + c];
}

// ||e||^2 per code, exact f32.
__global__ __launch_bounds__(256) void hvq_prep_esq(
    const float* __restrict__ e0, const float* __restrict__ e1,
    const float* __restrict__ e2, float* __restrict__ esq)
{
  int g = blockIdx.x * 256 + threadIdx.x;   // 0 .. 3*1024-1
  int l = g >> 10;
  int code = g & 1023;
  const float* e = (l == 0) ? e0 : ((l == 1) ? e1 : e2);
  const float4* p = (const float4*)(e + code * CDIM);
  float s = 0.f;
  #pragma unroll 4
  for (int i = 0; i < CDIM / 4; ++i) {
    float4 v = p[i];
    s += v.x * v.x + v.y * v.y + v.z * v.z + v.w * v.w;
  }
  esq[g] = s;
}

// ------------------------------------------------------------------
// One VQ layer.  2 waves / workgroup, each wave owns 32 rows (2 x 16-row
// tiles -> 2 accumulators per B fragment).  The 8 KB B code-tiles are
// double-buffered in LDS via GLOBAL_LOAD_ASYNC_TO_LDS_B128 and shared by
// both waves: 4x less L2 traffic than one-wave/16-row streaming, and no
// VGPR round-trip for staging.
__global__ __launch_bounds__(64) void hvq_layer(
    const float* xin, float* xout,               // may alias (tile-local)
    const float* __restrict__ emb,               // f32 [K,C]
    const _Float16* __restrict__ embT,           // prepped B tiles
    const float* __restrict__ esq,               // f32 [K]
    float* __restrict__ partial)                 // [NTILES]
{
  __shared__ __align__(16) _Float16 bbuf[2 * 4096];  // B double buffer, 16 KB
  __shared__ __align__(16) float xs[2 * 4096];       // per-wave x staging, 32 KB
  __shared__ float rminv[2 * 256];
  __shared__ int   rmini[2 * 256];
  __shared__ int   sel[2 * 16];

  const int tix  = threadIdx.x;                  // 0..63
  const int lane = tix & 31;
  const int w    = tix >> 5;                     // wave id 0/1
  const int tid0 = blockIdx.x * 4 + w * 2;       // first 16-row tile of wave

  float* xsw  = xs    + w * 4096;
  float* rmv  = rminv + w * 256;
  int*   rmi  = rmini + w * 256;
  int*   selw = sel   + w * 16;

  const unsigned bb_off  = lds_off(bbuf);
  const unsigned xsw_off = lds_off(xsw);

  const int m     = lane & 15;
  const int kbase = (lane & 16) >> 1;            // lanes>=16 -> +8

  // ---- build both A fragment sets (16x32 f16 chunks, contiguous k order) ----
  v16h a[2][8];
  #pragma unroll
  for (int tt = 0; tt < 2; ++tt) {
    const int tid = tid0 + tt;
    const int b   = tid >> 7;
    const int t0  = (tid & 127) << 4;
    const size_t xb = ((size_t)b * CDIM) * TLEN + (size_t)t0;
    for (int it = 0; it < 32; ++it) {
      int j4  = it * 32 + lane;                  // 0..1023 float4 slots
      int c   = j4 >> 2;
      int tof = (j4 & 3) << 2;
      async_ld_b128(xsw_off + (unsigned)(c * 16 + tof) * 4,
                    xin + xb + (size_t)c * TLEN + tof);
    }
    wait_async0();
    __syncthreads();                             // uniform across both waves
    #pragma unroll
    for (int kc = 0; kc < 8; ++kc) {
      #pragma unroll
      for (int h = 0; h < 16; ++h) {
        int K = h + 2 * kbase;                   // contiguous k = 16*hi + h (matches embT B)
        a[tt][kc][h] = (_Float16)xsw[(kc * 32 + K) * 16 + m];
      }
    }
    __syncthreads();
  }

  // ---- prologue: async-stage B tile 0 into LDS ----
  #pragma unroll
  for (int i = 0; i < 8; ++i)
    async_ld_b128(bb_off + (unsigned)(i * 64 + tix) * 16,
                  (const char*)embT + (size_t)(i * 64 + tix) * 16);
  wait_async0();
  __syncthreads();

  // ---- 64 code tiles: LDS-shared B, 16 WMMAs per tile per wave ----
  float minv[2][8];
  int   mini[2][8];
  #pragma unroll
  for (int d = 0; d < 8; ++d) {
    minv[0][d] = 3.0e38f; mini[0][d] = 0;
    minv[1][d] = 3.0e38f; mini[1][d] = 0;
  }

  for (int n = 0; n < 64; ++n) {
    if (n < 63) {                                // async-fetch next tile into other buffer
      const char* src = (const char*)embT + (size_t)(n + 1) * 8192;
      unsigned    dst = bb_off + (unsigned)(((n + 1) & 1) * 8192);
      #pragma unroll
      for (int i = 0; i < 8; ++i)
        async_ld_b128(dst + (unsigned)(i * 64 + tix) * 16,
                      src + (size_t)(i * 64 + tix) * 16);
    }
    if (n < 62)
      __builtin_prefetch((const void*)((const char*)embT + (size_t)(n + 2) * 8192 + tix * 16), 0, 0);

    const _Float16* bbp = bbuf + (size_t)(n & 1) * 4096;
    v8f acc0 = {}, acc1 = {};
    #pragma unroll
    for (int kc = 0; kc < 8; ++kc) {
      v16h bv = *(const v16h*)(bbp + kc * 512 + lane * 16);  // ds_load_b128 x2
      acc0 = __builtin_amdgcn_wmma_f32_16x16x32_f16(
          false, a[0][kc], false, bv, (short)0, acc0, false, false);
      acc1 = __builtin_amdgcn_wmma_f32_16x16x32_f16(
          false, a[1][kc], false, bv, (short)0, acc1, false, false);
    }

    int   code = n * 16 + m;
    float e2   = esq[code];
    #pragma unroll
    for (int d = 0; d < 8; ++d) {
      float s0 = e2 - 2.0f * acc0[d];            // ||e||^2 - 2 f.e
      bool  l0 = s0 < minv[0][d];
      minv[0][d] = l0 ? s0 : minv[0][d];
      mini[0][d] = l0 ? code : mini[0][d];
      float s1 = e2 - 2.0f * acc1[d];
      bool  l1 = s1 < minv[1][d];
      minv[1][d] = l1 ? s1 : minv[1][d];
      mini[1][d] = l1 ? code : mini[1][d];
    }

    if (n < 63) {
      wait_async0();                             // this wave's async LDS writes done
      __syncthreads();                           // both waves' writes visible
    }
  }

  // ---- per-wave epilogue: argmin finalize + gather + loss ----
  const int hi = lane >> 4;
  #pragma unroll
  for (int tt = 0; tt < 2; ++tt) {
    const int tid = tid0 + tt;
    const int b   = tid >> 7;
    const int t0  = (tid & 127) << 4;
    const size_t xb = ((size_t)b * CDIM) * TLEN + (size_t)t0;

    // cross-lane argmin per row (LDS ops are in-order within a wave)
    #pragma unroll
    for (int d = 0; d < 8; ++d) {
      int M = d + 8 * hi;                        // ISA C/D f32 row mapping
      rmv[M * 16 + m] = minv[tt][d];
      rmi[M * 16 + m] = mini[tt][d];
    }
    if (lane < 16) {
      float bv = rmv[lane * 16];
      int   bi = rmi[lane * 16];
      for (int s = 1; s < 16; ++s) {
        float v = rmv[lane * 16 + s];
        int   i = rmi[lane * 16 + s];
        if (v < bv || (v == bv && i < bi)) { bv = v; bi = i; }
      }
      selw[lane] = bi;
    }

    // gather codes, straight-through output, loss partial (x re-read hits L2)
    float sum = 0.f;
    for (int it = 0; it < 32; ++it) {
      int slot = it * 32 + lane;                 // 0..1023 float4 slots
      int mm   = slot & 15;
      int c    = (slot >> 4) << 2;
      int cd   = selw[mm];
      float4 e4 = *(const float4*)(emb + (size_t)cd * CDIM + c);
      float ev[4] = { e4.x, e4.y, e4.z, e4.w };
      #pragma unroll
      for (int p = 0; p < 4; ++p) {
        float xv = xin[xb + (size_t)(c + p) * TLEN + mm];
        float df = ev[p] - xv;                   // quant - x
        xout[xb + (size_t)(c + p) * TLEN + mm] = xv + df;  // x + sg(quant-x)
        sum += df * df;
      }
    }
    #pragma unroll
    for (int off = 16; off > 0; off >>= 1) sum += __shfl_xor(sum, off, 32);
    if (lane == 0) partial[tid] = sum;
  }
}

// ------------------------------------------------------------------
// Deterministic fixed-order reduction of 3*NTILES partials -> loss scalar.
__global__ __launch_bounds__(256) void hvq_reduce(
    const float* __restrict__ partial, float* __restrict__ loss)
{
  __shared__ float s[256];
  float a = 0.f;
  for (int i = threadIdx.x; i < 3 * NTILES; i += 256) a += partial[i];
  s[threadIdx.x] = a;
  __syncthreads();
  for (int off = 128; off > 0; off >>= 1) {
    if (threadIdx.x < off) s[threadIdx.x] += s[threadIdx.x + off];
    __syncthreads();
  }
  if (threadIdx.x == 0) loss[0] = s[0] * (2.0f / (float)NELEM);
}

// ------------------------------------------------------------------
extern "C" void kernel_launch(void* const* d_in, const int* in_sizes, int n_in,
                              void* d_out, int out_size, void* d_ws, size_t ws_size,
                              hipStream_t stream) {
  (void)in_sizes; (void)n_in; (void)out_size; (void)ws_size;
  const float* x  = (const float*)d_in[0];
  const float* e0 = (const float*)d_in[1];
  const float* e1 = (const float*)d_in[2];
  const float* e2 = (const float*)d_in[3];
  float* out = (float*)d_out;                    // [NELEM] quant + [1] loss

  // workspace (~1.6 MB): embT f16 [3*262144], esq f32 [3*1024], partials f32 [3*2048]
  char* ws = (char*)d_ws;
  _Float16* embT = (_Float16*)ws;
  float*    esq  = (float*)(ws + (size_t)3 * 262144 * sizeof(_Float16));
  float*    part = esq + 3 * 1024;

  hvq_prep_embT<<<3072, 256, 0, stream>>>(e0, e1, e2, embT);
  hvq_prep_esq<<<12, 256, 0, stream>>>(e0, e1, e2, esq);

  // layer 1: d_in -> d_out; layers 2,3 run tile-locally in place on d_out
  hvq_layer<<<NTILES / 4, 64, 0, stream>>>(x,   out, e0, embT,            esq,        part);
  hvq_layer<<<NTILES / 4, 64, 0, stream>>>(out, out, e1, embT + 262144,   esq + 1024, part + NTILES);
  hvq_layer<<<NTILES / 4, 64, 0, stream>>>(out, out, e2, embT + 2*262144, esq + 2048, part + 2*NTILES);

  hvq_reduce<<<1, 256, 0, stream>>>(part, out + NELEM);
}
